// Dy_FPAM_Module_69389491634872
// MI455X (gfx1250) — compile-verified, matrix operand order checked by reading, not run on hardware
//
#include <hip/hip_runtime.h>

typedef __attribute__((ext_vector_type(16))) __bf16 v16bf;
typedef __attribute__((ext_vector_type(8)))  __bf16 v8bf;
typedef __attribute__((ext_vector_type(8)))  float  v8f;
typedef __attribute__((ext_vector_type(4)))  float  v4f;

#define kB 32
#define kC 64
#define kCQ 8

__device__ __forceinline__ v8f wmma_bf16(v16bf a, v16bf b, v8f c) {
  return __builtin_amdgcn_wmma_f32_16x16x32_bf16(false, a, false, b, (short)0, c,
                                                 false, false);
}

// ---------------------------------------------------------------------------
// Fragment loaders. CDNA5 WMMA 16-bit operand layout (ISA 7.12.2):
// per lane, the 16 bf16 elements cover K offsets {0..7}+16*{0,1} (+8 for the
// upper lane half). `base` already points at (row, k0); no bounds checks --
// caller clamps the row index (OOB rows/cols are never stored) and uses the
// _tail variant for a ragged final K step (zero-fill via selects, no branches).
// UNIT=true (compile-time) -> 128-bit vector loads (call sites 16B aligned).
// ---------------------------------------------------------------------------
template <bool UNIT, typename T>
__device__ __forceinline__ v16bf frag16(const T* __restrict__ base, long sK) {
  const int half = (threadIdx.x & 31) >> 4;
  v16bf f;
  if constexpr (UNIT) {
    const T* p0 = base + half * 8;
    const T* p1 = p0 + 16;
    if constexpr (sizeof(T) == 4) {
      const v4f r0 = ((const v4f*)p0)[0];
      const v4f r1 = ((const v4f*)p0)[1];
      const v4f r2 = ((const v4f*)p1)[0];
      const v4f r3 = ((const v4f*)p1)[1];
#pragma unroll
      for (int i = 0; i < 4; ++i) {
        f[i]      = (__bf16)r0[i];
        f[4 + i]  = (__bf16)r1[i];
        f[8 + i]  = (__bf16)r2[i];
        f[12 + i] = (__bf16)r3[i];
      }
    } else {
      const v8bf r0 = ((const v8bf*)p0)[0];
      const v8bf r1 = ((const v8bf*)p1)[0];
#pragma unroll
      for (int i = 0; i < 8; ++i) {
        f[i]     = r0[i];
        f[8 + i] = r1[i];
      }
    }
  } else {
#pragma unroll
    for (int v = 0; v < 8; ++v) {
      const long kb = (long)(((v < 4) ? (2 * v) : (16 + 2 * (v - 4))) + half * 8);
      f[2 * v]     = (__bf16)(float)base[kb * sK];
      f[2 * v + 1] = (__bf16)(float)base[(kb + 1) * sK];
    }
  }
  return f;
}

template <typename T>
__device__ __forceinline__ v16bf frag16_tail(const T* __restrict__ base, long sK,
                                             int kRem) {
  const int half = (threadIdx.x & 31) >> 4;
  v16bf f;
#pragma unroll
  for (int v = 0; v < 8; ++v) {
    const int kb = ((v < 4) ? (2 * v) : (16 + 2 * (v - 4))) + half * 8;
#pragma unroll
    for (int e = 0; e < 2; ++e) {
      const int k = kb + e;
      const int kc = (k < kRem) ? k : (kRem - 1);        // clamp, always load
      const float x = (float)base[(long)kc * sK];
      const float xv = (k < kRem) ? x : 0.0f;            // zero-fill via select
      f[2 * v + e] = (__bf16)xv;
    }
  }
  return f;
}

// ---------------------------------------------------------------------------
// Generic strided WMMA GEMM: C[b][m][n] = sum_s sum_k A[b,s,m,k] * B[b,s,k,n]
// (+ bias per-m or per-n, optional accumulate into C). One wave per 16x16 tile.
// AK1/BK1: compile-time "K-stride is 1" (vectorized fragment loads).
// ---------------------------------------------------------------------------
template <bool AK1, bool BK1, typename TA, typename TB, typename TC>
__global__ void gemm_wmma(const TA* __restrict__ A, const TB* __restrict__ Bm,
                          TC* __restrict__ Cm, const float* __restrict__ bias,
                          int biasMode, int accumulate,
                          int M, int N, int K, int nSlice,
                          long sAb, long sAs, long sAm, long sAk,
                          long sBb, long sBs, long sBk, long sBn,
                          long sCb, long sCm) {
  const int wpb = blockDim.x >> 5;
  const int nTile = blockIdx.x * wpb + (threadIdx.x >> 5);
  const int n0 = nTile * 16;
  if (n0 >= N) return;  // wave-uniform
  const int m0 = blockIdx.y * 16;
  const int b = blockIdx.z;
  const int lane = threadIdx.x & 31;
  const int l15 = lane & 15;
  const int mC = (m0 + l15 < M) ? (m0 + l15) : (M - 1);  // clamped row
  const int nC = (n0 + l15 < N) ? (n0 + l15) : (N - 1);  // clamped col
  const int K32 = K & ~31;
  v8f acc = {};
  for (int s = 0; s < nSlice; ++s) {
    const TA* aRow = A + (long)b * sAb + (long)s * sAs + (long)mC * sAm;
    const TB* bRow = Bm + (long)b * sBb + (long)s * sBs + (long)nC * sBn;
    int k0 = 0;
    for (; k0 < K32; k0 += 32) {
      v16bf af = frag16<AK1, TA>(aRow + (long)k0 * sAk, sAk);
      v16bf bf = frag16<BK1, TB>(bRow + (long)k0 * sBk, sBk);
      acc = wmma_bf16(af, bf, acc);
    }
    if (k0 < K) {
      v16bf af = frag16_tail<TA>(aRow + (long)k0 * sAk, sAk, K - k0);
      v16bf bf = frag16_tail<TB>(bRow + (long)k0 * sBk, sBk, K - k0);
      acc = wmma_bf16(af, bf, acc);
    }
  }
  const int n = n0 + l15;
  const int hi = lane >> 4;
  if (n < N) {
#pragma unroll
    for (int r = 0; r < 8; ++r) {
      const int m = m0 + r + 8 * hi;
      if (m < M) {
        float v = acc[r];
        if (biasMode == 1) v += bias[m];
        else if (biasMode == 2) v += bias[n];
        TC* p = Cm + (long)b * sCb + (long)m * sCm + n;
        if (accumulate) *p = (TC)((float)*p + v);
        else            *p = (TC)v;
      }
    }
  }
}

// ---------------------------------------------------------------------------
// Dual-product combine: out = g1 * (GEMM with P1) + g2 * (GEMM with P2) + resid
// All operands bf16 with unit K-stride (vectorized).
// dualA=1: P1/P2 are the A operands (V), PY is shared B (attT)   [width]
// dualA=0: PY is shared A (attB), P1/P2 are the B operands (VT)  [height]
// ---------------------------------------------------------------------------
__global__ void combine_wmma(const __bf16* __restrict__ P1, const __bf16* __restrict__ P2,
                             const __bf16* __restrict__ PY, const float* __restrict__ resid,
                             float* __restrict__ Out, const float* __restrict__ gates,
                             int gbase, int dualA,
                             int M, int N, int K, int Cn,
                             long sP_b, long sP_c, long sY_b, long sY_c,
                             long sAm, long sBn,
                             long sO_b, long sO_c, long sO_m) {
  const int wpb = blockDim.x >> 5;
  const int nTile = blockIdx.x * wpb + (threadIdx.x >> 5);
  const int n0 = nTile * 16;
  if (n0 >= N) return;
  const int m0 = blockIdx.y * 16;
  const int bc = blockIdx.z;
  const int b = bc / Cn;
  const int c = bc % Cn;
  const int lane = threadIdx.x & 31;
  const int l15 = lane & 15;
  const int mC = (m0 + l15 < M) ? (m0 + l15) : (M - 1);
  const int nC = (n0 + l15 < N) ? (n0 + l15) : (N - 1);
  const int K32 = K & ~31;
  const __bf16* pBase1 = P1 + (long)b * sP_b + (long)c * sP_c;
  const __bf16* pBase2 = P2 + (long)b * sP_b + (long)c * sP_c;
  const __bf16* pyBase = PY + (long)b * sY_b + (long)c * sY_c;
  v8f acc1 = {}, acc2 = {};
  if (dualA) {
    const __bf16* a1Row = pBase1 + (long)mC * sAm;
    const __bf16* a2Row = pBase2 + (long)mC * sAm;
    const __bf16* bRow  = pyBase + (long)nC * sBn;
    int k0 = 0;
    for (; k0 < K32; k0 += 32) {
      v16bf bsh = frag16<true, __bf16>(bRow + k0, 1);
      v16bf a1 = frag16<true, __bf16>(a1Row + k0, 1);
      v16bf a2 = frag16<true, __bf16>(a2Row + k0, 1);
      acc1 = wmma_bf16(a1, bsh, acc1);
      acc2 = wmma_bf16(a2, bsh, acc2);
    }
    if (k0 < K) {
      v16bf bsh = frag16_tail<__bf16>(bRow + k0, 1, K - k0);
      v16bf a1 = frag16_tail<__bf16>(a1Row + k0, 1, K - k0);
      v16bf a2 = frag16_tail<__bf16>(a2Row + k0, 1, K - k0);
      acc1 = wmma_bf16(a1, bsh, acc1);
      acc2 = wmma_bf16(a2, bsh, acc2);
    }
  } else {
    const __bf16* aRow  = pyBase + (long)mC * sAm;
    const __bf16* b1Row = pBase1 + (long)nC * sBn;
    const __bf16* b2Row = pBase2 + (long)nC * sBn;
    int k0 = 0;
    for (; k0 < K32; k0 += 32) {
      v16bf ash = frag16<true, __bf16>(aRow + k0, 1);
      v16bf b1 = frag16<true, __bf16>(b1Row + k0, 1);
      v16bf b2 = frag16<true, __bf16>(b2Row + k0, 1);
      acc1 = wmma_bf16(ash, b1, acc1);
      acc2 = wmma_bf16(ash, b2, acc2);
    }
    if (k0 < K) {
      v16bf ash = frag16_tail<__bf16>(aRow + k0, 1, K - k0);
      v16bf b1 = frag16_tail<__bf16>(b1Row + k0, 1, K - k0);
      v16bf b2 = frag16_tail<__bf16>(b2Row + k0, 1, K - k0);
      acc1 = wmma_bf16(ash, b1, acc1);
      acc2 = wmma_bf16(ash, b2, acc2);
    }
  }
  const float g1 = gates[b * 4 + gbase];
  const float g2 = gates[b * 4 + gbase + 1];
  const int n = n0 + l15;
  const int hi = lane >> 4;
  if (n < N) {
#pragma unroll
    for (int r = 0; r < 8; ++r) {
      const int m = m0 + r + 8 * hi;
      if (m < M) {
        const long o = (long)b * sO_b + (long)c * sO_c + (long)m * sO_m + n;
        Out[o] = g1 * acc1[r] + g2 * acc2[r] + resid[o];
      }
    }
  }
}

// Q/K projection (C=64 -> CQ=8, both Q and K in one pass), output bf16.
__global__ void proj_qk_kernel(const float* __restrict__ x,
                               const float* __restrict__ qw, const float* __restrict__ qb,
                               const float* __restrict__ kw, const float* __restrict__ kb,
                               __bf16* __restrict__ Q, __bf16* __restrict__ K, int HW) {
  const int p = blockIdx.x * blockDim.x + threadIdx.x;
  const int b = blockIdx.y;
  if (p >= HW) return;
  const float* xb = x + (long)b * kC * HW + p;
  float aq[kCQ], ak[kCQ];
#pragma unroll
  for (int o = 0; o < kCQ; ++o) { aq[o] = qb[o]; ak[o] = kb[o]; }
  for (int c = 0; c < kC; ++c) {
    const float xv = xb[(long)c * HW];
#pragma unroll
    for (int o = 0; o < kCQ; ++o) {
      aq[o] = fmaf(qw[o * kC + c], xv, aq[o]);
      ak[o] = fmaf(kw[o * kC + c], xv, ak[o]);
    }
  }
  __bf16* Qb = Q + (long)b * kCQ * HW + p;
  __bf16* Kb = K + (long)b * kCQ * HW + p;
#pragma unroll
  for (int o = 0; o < kCQ; ++o) {
    Qb[(long)o * HW] = (__bf16)aq[o];
    Kb[(long)o * HW] = (__bf16)ak[o];
  }
}

// Transpose-pack input, channel-last: xT[b][p][c] = (bf16)x[b][c][p]
__global__ void xpose_kernel(const float* __restrict__ x, __bf16* __restrict__ xT,
                             int HW) {
  const int idx = blockIdx.x * blockDim.x + threadIdx.x;  // over kC*HW
  const int b = blockIdx.y;
  if (idx >= kC * HW) return;
  const int c = idx / HW;
  const int p = idx - c * HW;
  xT[(long)b * HW * kC + (long)p * kC + c] = (__bf16)x[(long)b * kC * HW + idx];
}

// Spatially-transposed channel-last: xT2[b][w*H+h][c] = (bf16)x[b][c][h*W+w]
__global__ void xpose2_kernel(const float* __restrict__ x, __bf16* __restrict__ xT2,
                              int H, int W) {
  const int HW = H * W;
  const int idx = blockIdx.x * blockDim.x + threadIdx.x;  // over kC*HW
  const int b = blockIdx.y;
  if (idx >= kC * HW) return;
  const int c = idx / HW;
  const int p = idx - c * HW;      // p = h*W + w
  const int h = p / W;
  const int w = p - h * W;
  const int pp = w * H + h;
  xT2[(long)b * HW * kC + (long)pp * kC + c] = (__bf16)x[(long)b * kC * HW + idx];
}

// Pack softmaxed attention into bf16, both row-major and transposed.
__global__ void attpack_kernel(const float* __restrict__ att, __bf16* __restrict__ attB,
                               __bf16* __restrict__ attT, int L) {
  const int idx = blockIdx.x * blockDim.x + threadIdx.x;  // over L*L
  const int b = blockIdx.y;
  const int L2 = L * L;
  if (idx >= L2) return;
  const int i = idx / L;
  const int j = idx - i * L;
  const __bf16 v = (__bf16)att[(long)b * L2 + idx];
  attB[(long)b * L2 + idx] = v;
  attT[(long)b * L2 + (long)j * L + i] = v;
}

// Row softmax, one wave per row of length L.
__global__ void softmax_rows(float* __restrict__ att, int L) {
  float* p = att + (long)blockIdx.x * L;
  const int lane = threadIdx.x;
  float mx = -1e30f;
  for (int j = lane; j < L; j += 32) mx = fmaxf(mx, p[j]);
#pragma unroll
  for (int off = 16; off > 0; off >>= 1) mx = fmaxf(mx, __shfl_xor(mx, off, 32));
  float sum = 0.0f;
  for (int j = lane; j < L; j += 32) { float e = __expf(p[j] - mx); p[j] = e; sum += e; }
#pragma unroll
  for (int off = 16; off > 0; off >>= 1) sum += __shfl_xor(sum, off, 32);
  const float inv = 1.0f / sum;
  for (int j = lane; j < L; j += 32) p[j] *= inv;
}

// Spatial mean of V[b][c][:] (bf16) -> avg[b*64+c] (fp32).
__global__ void mean_kernel(const __bf16* __restrict__ V, float* __restrict__ avg, int HW) {
  const long bc = blockIdx.x;
  const int lane = threadIdx.x;
  float s = 0.0f;
  const __bf16* p = V + bc * HW;
  for (int i = lane; i < HW; i += 32) s += (float)p[i];
#pragma unroll
  for (int off = 16; off > 0; off >>= 1) s += __shfl_xor(s, off, 32);
  if (lane == 0) avg[bc] = s / (float)HW;
}

// gates[b*4+gi] = sigmoid(dot(avg, gate_w[g0+gi])); gi even->pc avg, odd->rgb avg
__global__ void gates_kernel(const float* __restrict__ avgP, const float* __restrict__ avgR,
                             const float* __restrict__ gate_w, int g0,
                             float* __restrict__ gates) {
  const int b = blockIdx.x, gi = blockIdx.y;
  const float* avg = (gi & 1) ? avgR : avgP;
  const float* w = gate_w + (g0 + gi) * kC;
  const int lane = threadIdx.x;
  float s = avg[b * kC + lane] * w[lane] + avg[b * kC + lane + 32] * w[lane + 32];
#pragma unroll
  for (int off = 16; off > 0; off >>= 1) s += __shfl_xor(s, off, 32);
  if (lane == 0) gates[b * 4 + gi] = 1.0f / (1.0f + __expf(-s));
}

static inline int cdiv(int a, int b) { return (a + b - 1) / b; }

static void run_stream(const float* x_pc, const float* x_rgb, int H, int W,
                       int iw_pc, int iw_rgb, int ih_pc, int ih_rgb,
                       const float* fw_w, const float* fw_b,
                       const float* fh_w, const float* fh_b, int g0,
                       const float* qw, const float* qb,
                       const float* kw, const float* kb,
                       const float* vw, const float* vb,
                       const float* gate_w,
                       float* out_w, float* out_h,
                       char* ws, hipStream_t stream) {
  const int HW = H * W;
  const long QK_ELEMS = (long)kB * kCQ * 4096;   // bf16
  const long E_ELEMS  = (long)kB * 4096;         // f32 / bf16
  const long V_ELEMS  = (long)kB * kC * 4096;    // bf16
  __bf16* Qpc  = (__bf16*)ws;
  __bf16* Kpc  = Qpc + QK_ELEMS;
  __bf16* Qrgb = Kpc + QK_ELEMS;
  __bf16* Krgb = Qrgb + QK_ELEMS;
  float*  ePC  = (float*)(Krgb + QK_ELEMS);
  float*  eRGB = ePC + E_ELEMS;
  float*  att  = eRGB + E_ELEMS;
  __bf16* attB = (__bf16*)(att + E_ELEMS);
  __bf16* attT = attB + E_ELEMS;
  __bf16* Vpc  = attT + E_ELEMS;
  __bf16* Vrgb = Vpc + V_ELEMS;
  __bf16* xTpc = Vrgb + V_ELEMS;
  __bf16* xTrgb = xTpc + V_ELEMS;
  float*  avgP = (float*)(xTrgb + V_ELEMS);
  float*  avgR = avgP + (long)kB * kC;
  float*  gates = avgR + (long)kB * kC;

  const dim3 blk(128);   // 4 wave32s
  const int L2 = W * W;  // == H*H (square maps)

  // Channel-last packed inputs for the width-phase V convs.
  {
    dim3 g(cdiv(kC * HW, 256), kB);
    xpose_kernel<<<g, 256, 0, stream>>>(x_pc, xTpc, HW);
    xpose_kernel<<<g, 256, 0, stream>>>(x_rgb, xTrgb, HW);
  }

  // ======== width attention ========
  {
    dim3 g(cdiv(HW, 256), kB);
    proj_qk_kernel<<<g, 256, 0, stream>>>(x_pc, qw + iw_pc * kCQ * kC, qb + iw_pc * kCQ,
                                          kw + iw_pc * kCQ * kC, kb + iw_pc * kCQ,
                                          Qpc, Kpc, HW);
    proj_qk_kernel<<<g, 256, 0, stream>>>(x_rgb, qw + iw_rgb * kCQ * kC, qb + iw_rgb * kCQ,
                                          kw + iw_rgb * kCQ * kC, kb + iw_rgb * kCQ,
                                          Qrgb, Krgb, HW);
  }
  { // e_w[b,i,j] = sum_{k=(c,h)} Q[b,k,i] * K[b,k,j]; K-dim = 8*H (%32==0)
    dim3 g(cdiv(cdiv(W, 16), 4), cdiv(W, 16), kB);
    gemm_wmma<false, false, __bf16, __bf16, float><<<g, blk, 0, stream>>>(
        Qpc, Kpc, ePC, nullptr, 0, 0, W, W, kCQ * H, 1,
        (long)kCQ * HW, 0, 1, W,
        (long)kCQ * HW, 0, W, 1,
        (long)W * W, W);
    gemm_wmma<false, false, __bf16, __bf16, float><<<g, blk, 0, stream>>>(
        Qrgb, Krgb, eRGB, nullptr, 0, 0, W, W, kCQ * H, 1,
        (long)kCQ * HW, 0, 1, W,
        (long)kCQ * HW, 0, W, 1,
        (long)W * W, W);
  }
  { // fused = e_pc @ W[:, :L2]^T + e_rgb @ W[:, L2:]^T + b   (M=batch=32)
    dim3 g(cdiv(cdiv(L2, 16), 4), cdiv(kB, 16), 1);
    gemm_wmma<true, true, float, float, float><<<g, blk, 0, stream>>>(
        ePC, fw_w, att, fw_b, 2, 0, kB, L2, L2, 1,
        0, 0, L2, 1,
        0, 0, 1, (long)2 * L2,
        0, L2);
    gemm_wmma<true, true, float, float, float><<<g, blk, 0, stream>>>(
        eRGB, fw_w + L2, att, nullptr, 0, 1, kB, L2, L2, 1,
        0, 0, L2, 1,
        0, 0, 1, (long)2 * L2,
        0, L2);
  }
  softmax_rows<<<kB * W, 32, 0, stream>>>(att, W);
  attpack_kernel<<<dim3(cdiv(L2, 256), kB), 256, 0, stream>>>(att, attB, attT, W);
  { // V[b][c][p] = vw[idx] @ x + vb  (per batch: M=64, N=HW, K=64)
    dim3 g(cdiv(cdiv(HW, 16), 4), cdiv(kC, 16), kB);
    gemm_wmma<true, true, float, __bf16, __bf16><<<g, blk, 0, stream>>>(
        vw + iw_pc * kC * kC, xTpc, Vpc, vb + iw_pc * kC, 1, 0, kC, HW, kC, 1,
        0, 0, kC, 1,
        (long)HW * kC, 0, 1, kC,
        (long)kC * HW, HW);
    gemm_wmma<true, true, float, __bf16, __bf16><<<g, blk, 0, stream>>>(
        vw + iw_rgb * kC * kC, xTrgb, Vrgb, vb + iw_rgb * kC, 1, 0, kC, HW, kC, 1,
        0, 0, kC, 1,
        (long)HW * kC, 0, 1, kC,
        (long)kC * HW, HW);
  }
  mean_kernel<<<kB * kC, 32, 0, stream>>>(Vpc, avgP, HW);
  mean_kernel<<<kB * kC, 32, 0, stream>>>(Vrgb, avgR, HW);
  gates_kernel<<<dim3(kB, 4), 32, 0, stream>>>(avgP, avgR, gate_w, g0, gates);
  { // out_w[b,c,h,j] = ga*(V_pc[h,:]@att[:,j]) + gb*(V_rgb...) + x_pc
    // A = V[h][w], shared B = attT[j][w]: all unit K-stride
    dim3 g(cdiv(cdiv(W, 16), 4), cdiv(H, 16), kB * kC);
    combine_wmma<<<g, blk, 0, stream>>>(
        Vpc, Vrgb, attT, x_pc, out_w, gates, 0, 1,
        H, W, W, kC,
        (long)kC * HW, HW,
        (long)W * W, 0,
        W, W,
        (long)kC * HW, HW, W);
  }

  // ======== height attention ========
  { // Spatially-transposed channel-last input (xT buffers are free now).
    dim3 g(cdiv(kC * HW, 256), kB);
    xpose2_kernel<<<g, 256, 0, stream>>>(x_pc, xTpc, H, W);
    xpose2_kernel<<<g, 256, 0, stream>>>(x_rgb, xTrgb, H, W);
  }
  {
    dim3 g(cdiv(HW, 256), kB);
    proj_qk_kernel<<<g, 256, 0, stream>>>(x_pc, qw + ih_pc * kCQ * kC, qb + ih_pc * kCQ,
                                          kw + ih_pc * kCQ * kC, kb + ih_pc * kCQ,
                                          Qpc, Kpc, HW);
    proj_qk_kernel<<<g, 256, 0, stream>>>(x_rgb, qw + ih_rgb * kCQ * kC, qb + ih_rgb * kCQ,
                                          kw + ih_rgb * kCQ * kC, kb + ih_rgb * kCQ,
                                          Qrgb, Krgb, HW);
  }
  { // e_h[b,i,j] = sum_c sum_w Q[b,c,i,w] * K[b,c,j,w]; 8 slices over c
    dim3 g(cdiv(cdiv(H, 16), 4), cdiv(H, 16), kB);
    gemm_wmma<true, true, __bf16, __bf16, float><<<g, blk, 0, stream>>>(
        Qpc, Kpc, ePC, nullptr, 0, 0, H, H, W, kCQ,
        (long)kCQ * HW, HW, W, 1,
        (long)kCQ * HW, HW, 1, W,
        (long)H * H, H);
    gemm_wmma<true, true, __bf16, __bf16, float><<<g, blk, 0, stream>>>(
        Qrgb, Krgb, eRGB, nullptr, 0, 0, H, H, W, kCQ,
        (long)kCQ * HW, HW, W, 1,
        (long)kCQ * HW, HW, 1, W,
        (long)H * H, H);
  }
  {
    dim3 g(cdiv(cdiv(L2, 16), 4), cdiv(kB, 16), 1);
    gemm_wmma<true, true, float, float, float><<<g, blk, 0, stream>>>(
        ePC, fh_w, att, fh_b, 2, 0, kB, L2, L2, 1,
        0, 0, L2, 1,
        0, 0, 1, (long)2 * L2,
        0, L2);
    gemm_wmma<true, true, float, float, float><<<g, blk, 0, stream>>>(
        eRGB, fh_w + L2, att, nullptr, 0, 1, kB, L2, L2, 1,
        0, 0, L2, 1,
        0, 0, 1, (long)2 * L2,
        0, L2);
  }
  softmax_rows<<<kB * H, 32, 0, stream>>>(att, H);
  attpack_kernel<<<dim3(cdiv(L2, 256), kB), 256, 0, stream>>>(att, attB, attT, H);
  { // VT[b][c][w*H+h] = height V conv on spatially-transposed input
    dim3 g(cdiv(cdiv(HW, 16), 4), cdiv(kC, 16), kB);
    gemm_wmma<true, true, float, __bf16, __bf16><<<g, blk, 0, stream>>>(
        vw + ih_pc * kC * kC, xTpc, Vpc, vb + ih_pc * kC, 1, 0, kC, HW, kC, 1,
        0, 0, kC, 1,
        (long)HW * kC, 0, 1, kC,
        (long)kC * HW, HW);
    gemm_wmma<true, true, float, __bf16, __bf16><<<g, blk, 0, stream>>>(
        vw + ih_rgb * kC * kC, xTrgb, Vrgb, vb + ih_rgb * kC, 1, 0, kC, HW, kC, 1,
        0, 0, kC, 1,
        (long)HW * kC, 0, 1, kC,
        (long)kC * HW, HW);
  }
  { // out_h[b,c,j,w] = gc*(att[j,:]@V_pc[:,w]) + gd*(att@V_rgb) + x_pc
    // shared A = attB[j][h], B = VT[c][w*H+h]: all unit K-stride
    dim3 g(cdiv(cdiv(W, 16), 4), cdiv(H, 16), kB * kC);
    combine_wmma<<<g, blk, 0, stream>>>(
        Vpc, Vrgb, attB, x_pc, out_h, gates, 2, 0,
        H, W, H, kC,
        (long)kC * HW, HW,
        (long)H * H, 0,
        H, H,
        (long)kC * HW, HW, W);
  }
}

extern "C" void kernel_launch(void* const* d_in, const int* in_sizes, int n_in,
                              void* d_out, int out_size, void* d_ws, size_t ws_size,
                              hipStream_t stream) {
  (void)in_sizes; (void)n_in; (void)out_size;
  const float* rgb_t = (const float*)d_in[0];
  const float* rgb_s = (const float*)d_in[1];
  const float* x_t   = (const float*)d_in[2];
  const float* x_s   = (const float*)d_in[3];
  const float* qw    = (const float*)d_in[4];
  const float* qb    = (const float*)d_in[5];
  const float* kw    = (const float*)d_in[6];
  const float* kb    = (const float*)d_in[7];
  const float* vw    = (const float*)d_in[8];
  const float* vb    = (const float*)d_in[9];
  const float* ftw_w = (const float*)d_in[10];
  const float* ftw_b = (const float*)d_in[11];
  const float* fth_w = (const float*)d_in[12];
  const float* fth_b = (const float*)d_in[13];
  const float* fsw_w = (const float*)d_in[14];
  const float* fsw_b = (const float*)d_in[15];
  const float* fsh_w = (const float*)d_in[16];
  const float* fsh_b = (const float*)d_in[17];
  const float* gate_w = (const float*)d_in[18];
  float* out = (float*)d_out;
  char* ws = (char*)d_ws;

  // workspace: 4 bf16 Q/K + 3 f32 e/att + 2 bf16 att packs + 4 bf16 V/xT + small
  const size_t need =
      4 * (size_t)kB * kCQ * 4096 * 2 +
      3 * (size_t)kB * 4096 * 4 +
      2 * (size_t)kB * 4096 * 2 +
      4 * (size_t)kB * kC * 4096 * 2 +
      (2 * (size_t)kB * kC + (size_t)kB * 4) * 4;
  if (ws_size < need) return;

  const long T_ELEMS = (long)kB * kC * 40 * 40;
  const long S_ELEMS = (long)kB * kC * 64 * 64;
  float* t_out_w = out;
  float* t_out_h = out + T_ELEMS;
  float* s_out_w = out + 2 * T_ELEMS;
  float* s_out_h = out + 2 * T_ELEMS + S_ELEMS;

  run_stream(x_t, rgb_t, 40, 40, 0, 1, 2, 3, ftw_w, ftw_b, fth_w, fth_b, 0,
             qw, qb, kw, kb, vw, vb, gate_w, t_out_w, t_out_h, ws, stream);
  run_stream(x_s, rgb_s, 64, 64, 4, 5, 6, 7, fsw_w, fsw_b, fsh_w, fsh_b, 4,
             qw, qb, kw, kb, vw, vb, gate_w, s_out_w, s_out_h, ws, stream);
}
